// RSMLayer_26302379721234
// MI455X (gfx1250) — matrix operand order repaired and checked by reading, not compiled, not run on hardware
//
#include <hip/hip_runtime.h>
#include <math.h>

// ---------------------------------------------------------------------------
// RSM layer fused pipeline for gfx1250 (MI455X), wave32 + WMMA f16->f32.
// Single-pass Wb stream (BM=256 covers the whole batch), double-buffered LDS
// with register-staged prefetch and global_prefetch on the weight stream.
// ---------------------------------------------------------------------------

typedef __attribute__((ext_vector_type(16))) _Float16 v16h;
typedef __attribute__((ext_vector_type(8)))  _Float16 v8h;
typedef __attribute__((ext_vector_type(4)))  _Float16 v4h;
typedef __attribute__((ext_vector_type(8)))  float    v8f;

#define M_GROUPS 1024
#define N_CELLS  8
#define TOTAL    8192
#define K_GRPS   128
#define D_IN     2048
#define D_OUT    2048
#define BSZ      256

#define BM 256
#define BN 64
#define BK 32
#define SA 40   // LDS row stride in halves: 80 B = 5*16 B (alignment-safe, skews banks)

enum { EPI_ZA = 0, EPI_SIGMA = 1, EPI_PRED = 2 };

__device__ __forceinline__ unsigned int f2key(float f) {
    unsigned int u = __float_as_uint(f);
    return (u & 0x80000000u) ? ~u : (u | 0x80000000u);   // monotonic total order
}
__device__ __forceinline__ float key2f(unsigned int k) {
    unsigned int u = (k & 0x80000000u) ? (k ^ 0x80000000u) : ~k;
    return __uint_as_float(u);
}

__global__ void init_min_kernel(unsigned int* minKey) {
    *minKey = 0xFFFFFFFFu;
}

// C[M,N] = A[M,K] * B[K,N] (f32 row-major in global, f16 in LDS),
// f32 accumulate via v_wmma_f32_16x16x32_f16. M == BM == 256 (one block row).
// EPI_ZA   : out = acc + bias[n]
// EPI_SIGMA: out = acc + bias[n] + zA[m, n/8]; global min via atomicMin(key)
// EPI_PRED : out = acc + bias[n]
template <int EPI>
__global__ __launch_bounds__(256)
void gemm_wmma(const float* __restrict__ A, const float* __restrict__ B,
               float* __restrict__ Out, const float* __restrict__ bias,
               const float* __restrict__ zA, unsigned int* __restrict__ minKey,
               int N, int K)
{
    __shared__ __align__(16) _Float16 sA[2][BM * SA];   // row-major [m][k]
    __shared__ __align__(16) _Float16 sB[2][BN * SA];   // N-major  [n][k]
    __shared__ float red[256];

    const int tid   = threadIdx.x;
    const int lane  = tid & 31;
    const int wave  = tid >> 5;        // 0..7
    const int wm    = wave >> 1;       // 0..3  (64 rows each)
    const int wn    = wave & 1;        // 0..1  (32 cols each)
    const int lhalf = lane >> 4;       // 0/1
    const int l16   = lane & 15;
    const int bM    = blockIdx.y * BM; // 0 (single M tile)
    const int bN    = blockIdx.x * BN;

    v8f acc[4][2];
    #pragma unroll
    for (int mt = 0; mt < 4; ++mt)
        #pragma unroll
        for (int nt = 0; nt < 2; ++nt)
            #pragma unroll
            for (int e = 0; e < 8; ++e) acc[mt][nt][e] = 0.0f;

    float4 pa[8];   // A: 256x32 tile -> 8 float4 / thread
    float4 pb[2];   // B: 32x64 tile  -> 2 float4 / thread

    auto loadRegs = [&](int k0) {
        #pragma unroll
        for (int i = 0; i < 8; ++i) {
            int lin = tid + 256 * i;            // 0..2047
            int r = lin >> 3, c4 = lin & 7;     // 256 rows x 8 float4
            pa[i] = *(const float4*)(A + (size_t)(bM + r) * K + k0 + c4 * 4);
        }
        #pragma unroll
        for (int i = 0; i < 2; ++i) {
            int lin = tid + 256 * i;            // 0..511
            int kr = lin >> 4, c4 = lin & 15;   // 32 rows x 16 float4
            pb[i] = *(const float4*)(B + (size_t)(k0 + kr) * N + bN + c4 * 4);
        }
    };
    auto storeLds = [&](int buf) {
        #pragma unroll
        for (int i = 0; i < 8; ++i) {
            int lin = tid + 256 * i;
            int r = lin >> 3, c4 = lin & 7;
            v4h h = { (_Float16)pa[i].x, (_Float16)pa[i].y,
                      (_Float16)pa[i].z, (_Float16)pa[i].w };
            *(v4h*)&sA[buf][r * SA + c4 * 4] = h;
        }
        #pragma unroll
        for (int i = 0; i < 2; ++i) {
            int lin = tid + 256 * i;
            int kr = lin >> 4, c4 = lin & 15;
            int n = c4 * 4;                     // transpose to [n][k]
            sB[buf][(n + 0) * SA + kr] = (_Float16)pb[i].x;
            sB[buf][(n + 1) * SA + kr] = (_Float16)pb[i].y;
            sB[buf][(n + 2) * SA + kr] = (_Float16)pb[i].z;
            sB[buf][(n + 3) * SA + kr] = (_Float16)pb[i].w;
        }
    };
    auto compute = [&](int buf) {
        // B fragments (32x16): lane holds N=l16, K = lhalf*8+0..7, 16+lhalf*8+0..7
        v16h bf[2];
        #pragma unroll
        for (int nt = 0; nt < 2; ++nt) {
            int n = wn * 32 + nt * 16 + l16;
            v8h lo = *(const v8h*)&sB[buf][n * SA + lhalf * 8];
            v8h hi = *(const v8h*)&sB[buf][n * SA + lhalf * 8 + 16];
            bf[nt] = __builtin_shufflevector(lo, hi,
                     0,1,2,3,4,5,6,7,8,9,10,11,12,13,14,15);
        }
        // A fragments (16x32): lane holds M=l16,
        // VGPR0-3: K = lhalf*8+0..7 ; VGPR4-7: K = 16+lhalf*8+0..7
        #pragma unroll
        for (int mt = 0; mt < 4; ++mt) {
            int r = wm * 64 + mt * 16 + l16;
            v8h lo = *(const v8h*)&sA[buf][r * SA + lhalf * 8];
            v8h hi = *(const v8h*)&sA[buf][r * SA + lhalf * 8 + 16];
            v16h af = __builtin_shufflevector(lo, hi,
                      0,1,2,3,4,5,6,7,8,9,10,11,12,13,14,15);
            #pragma unroll
            for (int nt = 0; nt < 2; ++nt)
                acc[mt][nt] = __builtin_amdgcn_wmma_f32_16x16x32_f16(
                    false, af, false, bf[nt], (short)0, acc[mt][nt],
                    false, false);
        }
    };

    const int T = K / BK;
    loadRegs(0);
    storeLds(0);
    __syncthreads();
    for (int it = 0; it < T; ++it) {
        int nxt = it + 1;
        if (nxt < T) {
            loadRegs(nxt * BK);                  // overlap with compute below
            if (nxt + 4 < T) {                   // L2 prefetch of weight stream
                int lin = tid;                   // one cacheline probe / thread
                int kr = lin >> 4, c4 = lin & 15;
                const float* pf =
                    B + (size_t)((nxt + 4) * BK + kr) * N + bN + c4 * 4;
                __builtin_prefetch(pf, 0, 1);
            }
        }
        compute(it & 1);
        if (nxt < T) storeLds(nxt & 1);
        __syncthreads();
    }

    // --- Epilogue: C layout lanes 0-15 -> M=i, lanes 16-31 -> M=8+i; N=l16 ---
    float lmin = 3.402823466e38f;
    #pragma unroll
    for (int mt = 0; mt < 4; ++mt) {
        #pragma unroll
        for (int nt = 0; nt < 2; ++nt) {
            #pragma unroll
            for (int i = 0; i < 8; ++i) {
                int r = bM + wm * 64 + mt * 16 + lhalf * 8 + i;
                int c = bN + wn * 32 + nt * 16 + l16;
                float v = acc[mt][nt][i] + bias[c];
                if (EPI == EPI_SIGMA) {
                    v += zA[(size_t)r * M_GROUPS + (c >> 3)];
                    lmin = fminf(lmin, v);
                }
                Out[(size_t)r * N + c] = v;
            }
        }
    }
    if (EPI == EPI_SIGMA) {
        red[tid] = lmin;
        __syncthreads();
        for (int s = 128; s > 0; s >>= 1) {
            if (tid < s) red[tid] = fminf(red[tid], red[tid + s]);
            __syncthreads();
        }
        if (tid == 0) atomicMin(minKey, f2key(red[0]));
    }
}

// One workgroup (256 threads) per batch row: argmax cell per group,
// top-128 group selection (bitonic sort of 1024 lambdas), fused outputs.
__global__ __launch_bounds__(256)
void select_kernel(const float* __restrict__ sigma, const float* __restrict__ phi,
                   const float* __restrict__ psi, const unsigned int* __restrict__ minKey,
                   float* __restrict__ out_xb, float* __restrict__ out_phi,
                   float* __restrict__ out_psi, float* __restrict__ decode_in)
{
    __shared__ float lam[M_GROUPS];
    __shared__ int   wc[M_GROUPS];
    __shared__ float srt[M_GROUPS];
    __shared__ unsigned char sel[M_GROUPS];
    __shared__ int   cnt[256];

    const int b   = blockIdx.x;
    const int tid = threadIdx.x;
    const float gmin = key2f(*minKey);
    const float* sg = sigma + (size_t)b * TOTAL;
    const float* ph = phi   + (size_t)b * TOTAL;

    // per-group max of pi (first-index tiebreak, matching top_k k=1)
    for (int g = tid; g < M_GROUPS; g += 256) {
        float best = -3.402823466e38f; int bi = 0;
        #pragma unroll
        for (int c = 0; c < N_CELLS; ++c) {
            float s = sg[g * N_CELLS + c];
            float p = (1.0f - ph[g * N_CELLS + c]) * (s - gmin + 1.0f);
            if (p > best) { best = p; bi = c; }
        }
        lam[g] = best; wc[g] = bi; srt[g] = best;
    }
    __syncthreads();

    // bitonic sort ascending; 128th largest at index 1024-128
    for (int kk = 2; kk <= M_GROUPS; kk <<= 1) {
        for (int j = kk >> 1; j > 0; j >>= 1) {
            for (int t = tid; t < M_GROUPS; t += 256) {
                int ixj = t ^ j;
                if (ixj > t) {
                    bool asc = ((t & kk) == 0);
                    float a = srt[t], c = srt[ixj];
                    if ((a > c) == asc) { srt[t] = c; srt[ixj] = a; }
                }
            }
            __syncthreads();
        }
    }
    float thr = srt[M_GROUPS - K_GRPS];

    // exact-k tie handling in index order (matches jax.lax.top_k)
    int lc = 0;
    for (int g = tid; g < M_GROUPS; g += 256) if (lam[g] > thr) ++lc;
    cnt[tid] = lc;
    __syncthreads();
    for (int s = 128; s > 0; s >>= 1) {
        if (tid < s) cnt[tid] += cnt[tid + s];
        __syncthreads();
    }
    if (tid == 0) {
        int quota = K_GRPS - cnt[0];
        int taken = 0;
        for (int g = 0; g < M_GROUPS; ++g) {
            bool s_ = lam[g] > thr;
            if (!s_ && lam[g] == thr && taken < quota) { s_ = true; ++taken; }
            sel[g] = (unsigned char)s_;
        }
    }
    __syncthreads();

    // fused outputs
    for (int g = tid; g < M_GROUPS; g += 256) {
        bool sl = sel[g] != 0;
        int  bi = wc[g];
        float dmax = 0.0f;
        #pragma unroll
        for (int c = 0; c < N_CELLS; ++c) {
            size_t idx = (size_t)b * TOTAL + g * N_CELLS + c;
            float s = sigma[idx];
            float y = (sl && c == bi) ? tanhf(s) : 0.0f;
            float pn = fmaxf(phi[idx] * 0.5f, y);   // GAMMA = 0.5
            float qn = fmaxf(psi[idx] * 0.5f, y);   // EPS   = 0.5
            out_phi[idx] = pn;
            out_psi[idx] = qn;
            out_xb[idx]  = qn;                      // PRED_GAIN = 1, no norm
            if (sl && c == bi) dmax = fmaxf(y, 0.0f);
        }
        decode_in[(size_t)b * M_GROUPS + g] = dmax;
    }
}

extern "C" void kernel_launch(void* const* d_in, const int* in_sizes, int n_in,
                              void* d_out, int out_size, void* d_ws, size_t ws_size,
                              hipStream_t stream)
{
    const float* x_a = (const float*)d_in[0];
    const float* x_b = (const float*)d_in[1];
    const float* phi = (const float*)d_in[2];
    const float* psi = (const float*)d_in[3];
    const float* Wa  = (const float*)d_in[4];
    const float* ba  = (const float*)d_in[5];
    const float* Wb  = (const float*)d_in[6];
    const float* bb  = (const float*)d_in[7];
    const float* Wd  = (const float*)d_in[8];
    const float* bd  = (const float*)d_in[9];
    (void)in_sizes; (void)n_in; (void)out_size; (void)ws_size;

    // workspace layout (floats): sigma[256*8192] | zA[256*1024] | dec[256*1024] | minKey
    float* ws    = (float*)d_ws;
    float* sigma = ws;                         // 2,097,152 f
    float* zA    = ws + 2097152;               // 262,144 f
    float* dec   = ws + 2097152 + 262144;      // 262,144 f
    unsigned int* minKey = (unsigned int*)(ws + 2097152 + 2 * 262144);

    float* pred = (float*)d_out;               // [256,2048]
    float* oxb  = pred + BSZ * D_OUT;          // [256,8192]
    float* ophi = oxb  + BSZ * TOTAL;          // [256,8192]
    float* opsi = ophi + BSZ * TOTAL;          // [256,8192]

    dim3 blk(256);

    init_min_kernel<<<dim3(1), dim3(1), 0, stream>>>(minKey);

    // z_a = x_a @ Wa + ba : M=256, N=1024, K=2048
    gemm_wmma<EPI_ZA><<<dim3(M_GROUPS / BN, BSZ / BM), blk, 0, stream>>>(
        x_a, Wa, zA, ba, nullptr, nullptr, M_GROUPS, D_IN);

    // sigma = x_b @ Wb + bb + repeat(z_a) : M=256, N=8192, K=8192 (+ global min)
    gemm_wmma<EPI_SIGMA><<<dim3(TOTAL / BN, BSZ / BM), blk, 0, stream>>>(
        x_b, Wb, sigma, bb, zA, minKey, TOTAL, TOTAL);

    // masks / tanh / phi,psi,x_b / decode vector
    select_kernel<<<dim3(BSZ), blk, 0, stream>>>(
        sigma, phi, psi, minKey, oxb, ophi, opsi, dec);

    // pred = decode @ Wd + bd : M=256, N=2048, K=1024
    gemm_wmma<EPI_PRED><<<dim3(D_OUT / BN, BSZ / BM), blk, 0, stream>>>(
        dec, Wd, pred, bd, nullptr, nullptr, D_OUT, M_GROUPS);
}
